// SlotAttention_15934328668767
// MI455X (gfx1250) — compile-verified, matrix operand order checked by reading, not run on hardware
//
#include <hip/hip_runtime.h>

#define B_ 32
#define N_ 2048
#define D_ 768
#define K_ 8
#define ITERS_ 3
#define HID_ 1536
#define G3D_ (3 * D_)
#define SCALE_ 0.35355339059327373f
#define EPS_ 1e-8f
#define LNEPS_ 1e-5f

typedef __bf16 bf16_t;
typedef __attribute__((ext_vector_type(16))) __bf16 bf16x16;
typedef __attribute__((ext_vector_type(8))) float floatx8;

__device__ __forceinline__ floatx8 zero8() {
  floatx8 z;
#pragma unroll
  for (int i = 0; i < 8; ++i) z[i] = 0.0f;
  return z;
}

__device__ __forceinline__ floatx8 wmma_bf16(bf16x16 a, bf16x16 b, floatx8 c) {
  return __builtin_amdgcn_wmma_f32_16x16x32_bf16(false, a, false, b, (short)0, c,
                                                 false, false);
}

// Fragment loader for an operand stored row-major [rows x lda], contracted over
// its columns. lane&15 selects row (row0+m). K striping per CDNA5 16-bit WMMA:
// lanes 0-15: K = {0..7, 16..23}+  ; lanes 16-31: +8.
__device__ __forceinline__ bf16x16 frag_rm(const bf16_t* base, int row0, int lda,
                                           int kk, int lane) {
  int m = lane & 15, hi = lane >> 4;
  const bf16_t* p = base + (size_t)(row0 + m) * lda + kk + hi * 8;
  bf16x16 f;
#pragma unroll
  for (int j = 0; j < 8; ++j) {
    int off = (j < 4) ? (2 * j) : (16 + 2 * (j - 4));
    f[2 * j] = p[off];
    f[2 * j + 1] = p[off + 1];
  }
  return f;
}

// Fragment loader for a B operand stored row-major [Kdim x ldb]; lane&15 selects
// output column col0+n, walk down the K dimension with stride ldb.
__device__ __forceinline__ bf16x16 frag_cm(const bf16_t* base, int col0, int ldb,
                                           int kk, int lane) {
  int n = lane & 15, hi = lane >> 4;
  const bf16_t* p = base + (size_t)(kk + hi * 8) * ldb + col0 + n;
  bf16x16 f;
#pragma unroll
  for (int i = 0; i < 16; ++i) {
    int krel = (i < 8) ? i : (i + 8);
    f[i] = p[(size_t)krel * ldb];
  }
  return f;
}

__device__ __forceinline__ float wave_sum(float s) {
#pragma unroll
  for (int off = 16; off > 0; off >>= 1) s += __shfl_xor(s, off, 32);
  return s;
}

// ---------------------------------------------------------------- converters
__global__ void cvt_bf16_kernel(const float* __restrict__ src,
                                bf16_t* __restrict__ dst, int n) {
  int i = blockIdx.x * 256 + threadIdx.x;
  if (i < n) dst[i] = (bf16_t)src[i];
}

__global__ void init_slots_kernel(const float* __restrict__ si,
                                  float* __restrict__ sf,
                                  bf16_t* __restrict__ sb) {
  int idx = blockIdx.x * 256 + threadIdx.x;  // B*K*D
  int d = idx % D_;
  int rk = idx / D_;
  int bb = rk / K_, k = rk % K_;
  float v = si[d];
  sf[idx] = v;
  sb[((size_t)bb * 16 + k) * D_ + d] = (bf16_t)v;
}

// ------------------------------------------------------------- LN statistics
__global__ void ln_stats_kernel(const float* __restrict__ x,
                                float* __restrict__ mean,
                                float* __restrict__ rstd) {
  int row = blockIdx.x * 8 + (threadIdx.x >> 5);
  int lane = threadIdx.x & 31;
  const float* p = x + (size_t)row * D_;
  float s = 0.f;
  for (int i = lane; i < D_; i += 32) s += p[i];
  s = wave_sum(s);
  float m = s / (float)D_;
  float v = 0.f;
  for (int i = lane; i < D_; i += 32) {
    float d = p[i] - m;
    v += d * d;
  }
  v = wave_sum(v);
  float r = rsqrtf(v / (float)D_ + LNEPS_);
  if (lane == 0) {
    mean[row] = m;
    rstd[row] = r;
  }
}

// LN of 8 small rows per batch -> bf16 [B,16,D] (rows 8..15 untouched)
__global__ void ln_rows_kernel(const float* __restrict__ src,
                               const float* __restrict__ g,
                               const float* __restrict__ bta,
                               bf16_t* __restrict__ dst) {
  int row = blockIdx.x * 8 + (threadIdx.x >> 5);  // 0..B*K-1
  int lane = threadIdx.x & 31;
  int bb = row / K_, k = row % K_;
  const float* p = src + (size_t)row * D_;
  float s = 0.f;
  for (int i = lane; i < D_; i += 32) s += p[i];
  s = wave_sum(s);
  float m = s / (float)D_;
  float v = 0.f;
  for (int i = lane; i < D_; i += 32) {
    float d = p[i] - m;
    v += d * d;
  }
  v = wave_sum(v);
  float r = rsqrtf(v / (float)D_ + LNEPS_);
  bf16_t* dp = dst + ((size_t)bb * 16 + k) * D_;
  for (int i = lane; i < D_; i += 32)
    dp[i] = (bf16_t)((p[i] - m) * r * g[i] + bta[i]);
}

// ---------------------------------------------- fused LN + k/v projections
__global__ void project_kv_kernel(const float* __restrict__ x,
                                  const float* __restrict__ mean,
                                  const float* __restrict__ rstd,
                                  const float* __restrict__ g,
                                  const float* __restrict__ bta,
                                  const bf16_t* __restrict__ wkb,
                                  const float* __restrict__ bk,
                                  const bf16_t* __restrict__ wvb,
                                  const float* __restrict__ bv,
                                  bf16_t* __restrict__ kout,
                                  bf16_t* __restrict__ vout) {
  __shared__ bf16_t ax[16 * D_];
  int blk = blockIdx.x;  // b*128 + ntile
  int bb = blk >> 7, nt = blk & 127;
  int row0 = nt * 16;
  const float* xb = x + ((size_t)bb * N_ + row0) * D_;
  const float* mb = mean + (size_t)bb * N_ + row0;
  const float* rb = rstd + (size_t)bb * N_ + row0;
  for (int idx = threadIdx.x; idx < 16 * D_; idx += 256) {
    int r = idx / D_, c = idx % D_;
    float val = (xb[(size_t)r * D_ + c] - mb[r]) * rb[r] * g[c] + bta[c];
    ax[idx] = (bf16_t)val;
  }
  __syncthreads();
  int wave = threadIdx.x >> 5, lane = threadIdx.x & 31;
  for (int ct = 0; ct < 6; ++ct) {
    int col0 = wave * 96 + ct * 16;
    floatx8 ak = zero8(), av = zero8();
    for (int kk = 0; kk < D_; kk += 32) {
      bf16x16 a = frag_rm(ax, 0, D_, kk, lane);
      bf16x16 fk = frag_rm(wkb, col0, D_, kk, lane);
      bf16x16 fv = frag_rm(wvb, col0, D_, kk, lane);
      ak = wmma_bf16(a, fk, ak);
      av = wmma_bf16(a, fv, av);
    }
    int n = lane & 15, hi = lane >> 4;
    size_t orow = (size_t)bb * N_ + row0;
#pragma unroll
    for (int r = 0; r < 8; ++r) {
      int m = r + hi * 8;
      size_t o = (orow + m) * D_ + col0 + n;
      kout[o] = (bf16_t)(ak[r] + bk[col0 + n]);
      vout[o] = (bf16_t)(av[r] + bv[col0 + n]);
    }
  }
}

// ------------------------------------------------------------ q projection
__global__ void gemm_q_kernel(const bf16_t* __restrict__ s,
                              const bf16_t* __restrict__ wqb,
                              const float* __restrict__ bq,
                              bf16_t* __restrict__ q) {
  int w = blockIdx.x * 8 + (threadIdx.x >> 5);
  int lane = threadIdx.x & 31;
  int bb = w / 48, ct = w % 48, col0 = ct * 16;
  const bf16_t* A = s + (size_t)bb * 16 * D_;
  floatx8 acc = zero8();
  for (int kk = 0; kk < D_; kk += 32)
    acc = wmma_bf16(frag_rm(A, 0, D_, kk, lane), frag_rm(wqb, col0, D_, kk, lane), acc);
  int n = lane & 15, hi = lane >> 4;
  bf16_t* Q = q + (size_t)bb * 16 * D_;
#pragma unroll
  for (int r = 0; r < 8; ++r)
    Q[(size_t)(r + 8 * hi) * D_ + col0 + n] = (bf16_t)(acc[r] + bq[col0 + n]);
}

// ------------------------------------------------------------------ logits
__global__ void gemm_logits_kernel(const bf16_t* __restrict__ kmat,
                                   const bf16_t* __restrict__ q,
                                   float* __restrict__ logits) {
  int w = blockIdx.x * 8 + (threadIdx.x >> 5);
  int lane = threadIdx.x & 31;
  int bb = w >> 7, mt = w & 127;  // N_/16 tiles
  const bf16_t* A = kmat + (size_t)bb * N_ * D_;
  const bf16_t* Q = q + (size_t)bb * 16 * D_;
  floatx8 acc = zero8();
  for (int kk = 0; kk < D_; kk += 32)
    acc = wmma_bf16(frag_rm(A, mt * 16, D_, kk, lane), frag_rm(Q, 0, D_, kk, lane), acc);
  int n = lane & 15, hi = lane >> 4;
  if (n < K_) {
    float* L = logits + ((size_t)bb * N_ + mt * 16) * K_;
#pragma unroll
    for (int r = 0; r < 8; ++r)
      L[(size_t)(r + 8 * hi) * K_ + n] = acc[r] * SCALE_;
  }
}

// --------------------------------------------- softmax over slots + colsums
__global__ void softmax_kernel(const float* __restrict__ logits,
                               float* __restrict__ attns_base, int it,
                               float* __restrict__ colsum) {
  __shared__ float ssum[K_];
  int row = blockIdx.x * 256 + threadIdx.x;  // B*N rows; blocks never straddle b
  int bb = row >> 11, n = row & (N_ - 1);
  if (threadIdx.x < K_) ssum[threadIdx.x] = 0.f;
  __syncthreads();
  const float* lp = logits + (size_t)row * K_;
  float v[K_];
  float mx = -1e30f;
#pragma unroll
  for (int i = 0; i < K_; ++i) {
    v[i] = lp[i];
    mx = fmaxf(mx, v[i]);
  }
  float s = 0.f;
#pragma unroll
  for (int i = 0; i < K_; ++i) {
    v[i] = __expf(v[i] - mx);
    s += v[i];
  }
  float inv = 1.f / s;
  float* ap = attns_base + ((size_t)bb * ITERS_ + it) * (size_t)N_ * K_ + (size_t)n * K_;
#pragma unroll
  for (int i = 0; i < K_; ++i) {
    float a = v[i] * inv + EPS_;
    ap[i] = a;
    atomicAdd(&ssum[i], a);
  }
  __syncthreads();
  if (threadIdx.x < K_) atomicAdd(&colsum[bb * K_ + threadIdx.x], ssum[threadIdx.x]);
}

// ---------------------------------------- renorm over inputs + bf16 transpose
__global__ void attn_norm_kernel(const float* __restrict__ attns_base, int it,
                                 const float* __restrict__ colsum,
                                 float* __restrict__ attn_out,
                                 bf16_t* __restrict__ attnT) {
  int row = blockIdx.x * 256 + threadIdx.x;
  int bb = row >> 11, n = row & (N_ - 1);
  const float* ap = attns_base + ((size_t)bb * ITERS_ + it) * (size_t)N_ * K_ + (size_t)n * K_;
#pragma unroll
  for (int i = 0; i < K_; ++i) {
    float a = ap[i] / colsum[bb * K_ + i];
    attn_out[(size_t)row * K_ + i] = a;
    attnT[((size_t)bb * 16 + i) * N_ + n] = (bf16_t)a;
  }
}

// ------------------------------------------------- updates = attn_out^T @ v
__global__ void gemm_updates_kernel(const bf16_t* __restrict__ attnT,
                                    const bf16_t* __restrict__ v,
                                    bf16_t* __restrict__ upd) {
  int w = blockIdx.x * 8 + (threadIdx.x >> 5);
  int lane = threadIdx.x & 31;
  int bb = w / 48, ct = w % 48, col0 = ct * 16;
  const bf16_t* A = attnT + (size_t)bb * 16 * N_;
  const bf16_t* V = v + (size_t)bb * N_ * D_;
  floatx8 acc = zero8();
  for (int kk = 0; kk < N_; kk += 32)
    acc = wmma_bf16(frag_rm(A, 0, N_, kk, lane), frag_cm(V, col0, D_, kk, lane), acc);
  int n = lane & 15, hi = lane >> 4;
  bf16_t* U = upd + (size_t)bb * 16 * D_;
#pragma unroll
  for (int r = 0; r < 8; ++r)
    U[(size_t)(r + 8 * hi) * D_ + col0 + n] = (bf16_t)acc[r];
}

// --------------------------------------------------------- GRU gate GEMMs
__global__ void gemm_gru_kernel(const bf16_t* __restrict__ upd,
                                const bf16_t* __restrict__ slots_bf,
                                const bf16_t* __restrict__ wihb,
                                const bf16_t* __restrict__ whhb,
                                const float* __restrict__ bih,
                                const float* __restrict__ bhh,
                                float* __restrict__ gi, float* __restrict__ gh) {
  int w = blockIdx.x * 8 + (threadIdx.x >> 5);
  int lane = threadIdx.x & 31;
  int bb = w / 144, ct = w % 144, col0 = ct * 16;
  const bf16_t* A1 = upd + (size_t)bb * 16 * D_;
  const bf16_t* A2 = slots_bf + (size_t)bb * 16 * D_;
  floatx8 ai = zero8(), ah = zero8();
  for (int kk = 0; kk < D_; kk += 32) {
    bf16x16 a1 = frag_rm(A1, 0, D_, kk, lane);
    bf16x16 a2 = frag_rm(A2, 0, D_, kk, lane);
    bf16x16 f1 = frag_rm(wihb, col0, D_, kk, lane);
    bf16x16 f2 = frag_rm(whhb, col0, D_, kk, lane);
    ai = wmma_bf16(a1, f1, ai);
    ah = wmma_bf16(a2, f2, ah);
  }
  int n = lane & 15, hi = lane >> 4;
  size_t gb = (size_t)bb * 16 * G3D_;
#pragma unroll
  for (int r = 0; r < 8; ++r) {
    size_t o = gb + (size_t)(r + 8 * hi) * G3D_ + col0 + n;
    gi[o] = ai[r] + bih[col0 + n];
    gh[o] = ah[r] + bhh[col0 + n];
  }
}

// ----------------------------------------------------------- GRU elementwise
__global__ void gru_elem_kernel(const float* __restrict__ gi,
                                const float* __restrict__ gh,
                                const float* __restrict__ slots_f,
                                float* __restrict__ h) {
  int idx = blockIdx.x * 256 + threadIdx.x;  // B*K*D
  int d = idx % D_;
  int rk = idx / D_;
  int bb = rk / K_, k = rk % K_;
  size_t base = ((size_t)bb * 16 + k) * G3D_ + d;
  float ir = gi[base], iz = gi[base + D_], in_ = gi[base + 2 * D_];
  float hr = gh[base], hz = gh[base + D_], hn = gh[base + 2 * D_];
  float r = 1.f / (1.f + __expf(-(ir + hr)));
  float z = 1.f / (1.f + __expf(-(iz + hz)));
  float nn = tanhf(in_ + r * hn);
  h[idx] = (1.f - z) * nn + z * slots_f[idx];
}

// ------------------------------------------------------------------- MLP 1
__global__ void gemm_mlp1_kernel(const bf16_t* __restrict__ m,
                                 const bf16_t* __restrict__ w1b,
                                 const float* __restrict__ b1f,
                                 bf16_t* __restrict__ hid) {
  int w = blockIdx.x * 8 + (threadIdx.x >> 5);
  int lane = threadIdx.x & 31;
  int bb = w / 96, ct = w % 96, col0 = ct * 16;
  const bf16_t* A = m + (size_t)bb * 16 * D_;
  floatx8 acc = zero8();
  for (int kk = 0; kk < D_; kk += 32)
    acc = wmma_bf16(frag_rm(A, 0, D_, kk, lane), frag_rm(w1b, col0, D_, kk, lane), acc);
  int n = lane & 15, hi = lane >> 4;
  bf16_t* H = hid + (size_t)bb * 16 * HID_;
#pragma unroll
  for (int r = 0; r < 8; ++r)
    H[(size_t)(r + 8 * hi) * HID_ + col0 + n] =
        (bf16_t)fmaxf(acc[r] + b1f[col0 + n], 0.f);
}

// -------------------------------------------------- MLP 2 + residual + slots
__global__ void gemm_mlp2_kernel(const bf16_t* __restrict__ hid,
                                 const bf16_t* __restrict__ w2b,
                                 const float* __restrict__ b2f,
                                 const float* __restrict__ h,
                                 float* __restrict__ slots_f,
                                 bf16_t* __restrict__ slots_bf) {
  int w = blockIdx.x * 8 + (threadIdx.x >> 5);
  int lane = threadIdx.x & 31;
  int bb = w / 48, ct = w % 48, col0 = ct * 16;
  const bf16_t* A = hid + (size_t)bb * 16 * HID_;
  floatx8 acc = zero8();
  for (int kk = 0; kk < HID_; kk += 32)
    acc = wmma_bf16(frag_rm(A, 0, HID_, kk, lane), frag_rm(w2b, col0, HID_, kk, lane), acc);
  int n = lane & 15, hi = lane >> 4;
#pragma unroll
  for (int r = 0; r < 8; ++r) {
    int mrow = r + 8 * hi;
    if (mrow < K_) {
      size_t o = ((size_t)bb * K_ + mrow) * D_ + col0 + n;
      float val = h[o] + acc[r] + b2f[col0 + n];
      slots_f[o] = val;
      slots_bf[((size_t)bb * 16 + mrow) * D_ + col0 + n] = (bf16_t)val;
    }
  }
}

extern "C" void kernel_launch(void* const* d_in, const int* in_sizes, int n_in,
                              void* d_out, int out_size, void* d_ws, size_t ws_size,
                              hipStream_t stream) {
  (void)in_sizes; (void)n_in; (void)out_size; (void)ws_size;
  const float* inputs     = (const float*)d_in[0];
  const float* slots_init = (const float*)d_in[1];
  const float* ln_in_g    = (const float*)d_in[2];
  const float* ln_in_b    = (const float*)d_in[3];
  const float* ln_slot_g  = (const float*)d_in[4];
  const float* ln_slot_b  = (const float*)d_in[5];
  const float* ln_mlp_g   = (const float*)d_in[6];
  const float* ln_mlp_b   = (const float*)d_in[7];
  const float* wq  = (const float*)d_in[8];  const float* bq   = (const float*)d_in[9];
  const float* wk  = (const float*)d_in[10]; const float* bk   = (const float*)d_in[11];
  const float* wv  = (const float*)d_in[12]; const float* bv   = (const float*)d_in[13];
  const float* wih = (const float*)d_in[14]; const float* bih  = (const float*)d_in[15];
  const float* whh = (const float*)d_in[16]; const float* bhh  = (const float*)d_in[17];
  const float* w1  = (const float*)d_in[18]; const float* b1   = (const float*)d_in[19];
  const float* w2  = (const float*)d_in[20]; const float* b2   = (const float*)d_in[21];

  float* out_slots = (float*)d_out;
  float* out_attn  = out_slots + (size_t)B_ * K_ * D_;
  float* out_attns = out_attn + (size_t)B_ * N_ * K_;

  char* base = (char*)d_ws;
  size_t off = 0;
  auto carve = [&](size_t bytes) -> void* {
    void* p = base + off;
    off += (bytes + 255) & ~(size_t)255;
    return p;
  };
  bf16_t* wq_bf  = (bf16_t*)carve((size_t)D_ * D_ * 2);
  bf16_t* wk_bf  = (bf16_t*)carve((size_t)D_ * D_ * 2);
  bf16_t* wv_bf  = (bf16_t*)carve((size_t)D_ * D_ * 2);
  bf16_t* wih_bf = (bf16_t*)carve((size_t)G3D_ * D_ * 2);
  bf16_t* whh_bf = (bf16_t*)carve((size_t)G3D_ * D_ * 2);
  bf16_t* w1_bf  = (bf16_t*)carve((size_t)HID_ * D_ * 2);
  bf16_t* w2_bf  = (bf16_t*)carve((size_t)D_ * HID_ * 2);
  float*  mean_  = (float*)carve((size_t)B_ * N_ * 4);
  float*  rstd_  = (float*)carve((size_t)B_ * N_ * 4);
  bf16_t* kbf    = (bf16_t*)carve((size_t)B_ * N_ * D_ * 2);
  bf16_t* vbf    = (bf16_t*)carve((size_t)B_ * N_ * D_ * 2);
  float*  slots_f  = (float*)carve((size_t)B_ * K_ * D_ * 4);
  bf16_t* slots_bf = (bf16_t*)carve((size_t)B_ * 16 * D_ * 2);
  bf16_t* s_bf     = (bf16_t*)carve((size_t)B_ * 16 * D_ * 2);
  bf16_t* q_bf     = (bf16_t*)carve((size_t)B_ * 16 * D_ * 2);
  bf16_t* upd_bf   = (bf16_t*)carve((size_t)B_ * 16 * D_ * 2);
  bf16_t* m_bf     = (bf16_t*)carve((size_t)B_ * 16 * D_ * 2);
  float*  logits   = (float*)carve((size_t)B_ * N_ * K_ * 4);
  float*  colsum   = (float*)carve((size_t)B_ * K_ * 4);
  bf16_t* attnT    = (bf16_t*)carve((size_t)B_ * 16 * N_ * 2);
  float*  gi       = (float*)carve((size_t)B_ * 16 * G3D_ * 4);
  float*  gh       = (float*)carve((size_t)B_ * 16 * G3D_ * 4);
  float*  h_f      = (float*)carve((size_t)B_ * K_ * D_ * 4);
  bf16_t* hid_bf   = (bf16_t*)carve((size_t)B_ * 16 * HID_ * 2);

  auto nb = [](int n) { return (n + 255) / 256; };
  cvt_bf16_kernel<<<nb(D_ * D_), 256, 0, stream>>>(wq, wq_bf, D_ * D_);
  cvt_bf16_kernel<<<nb(D_ * D_), 256, 0, stream>>>(wk, wk_bf, D_ * D_);
  cvt_bf16_kernel<<<nb(D_ * D_), 256, 0, stream>>>(wv, wv_bf, D_ * D_);
  cvt_bf16_kernel<<<nb(G3D_ * D_), 256, 0, stream>>>(wih, wih_bf, G3D_ * D_);
  cvt_bf16_kernel<<<nb(G3D_ * D_), 256, 0, stream>>>(whh, whh_bf, G3D_ * D_);
  cvt_bf16_kernel<<<nb(HID_ * D_), 256, 0, stream>>>(w1, w1_bf, HID_ * D_);
  cvt_bf16_kernel<<<nb(D_ * HID_), 256, 0, stream>>>(w2, w2_bf, D_ * HID_);

  init_slots_kernel<<<nb(B_ * K_ * D_), 256, 0, stream>>>(slots_init, slots_f, slots_bf);
  ln_stats_kernel<<<(B_ * N_) / 8, 256, 0, stream>>>(inputs, mean_, rstd_);
  project_kv_kernel<<<B_ * (N_ / 16), 256, 0, stream>>>(
      inputs, mean_, rstd_, ln_in_g, ln_in_b, wk_bf, bk, wv_bf, bv, kbf, vbf);

  for (int it = 0; it < ITERS_; ++it) {
    ln_rows_kernel<<<(B_ * K_) / 8, 256, 0, stream>>>(slots_f, ln_slot_g, ln_slot_b, s_bf);
    gemm_q_kernel<<<(B_ * 48) / 8, 256, 0, stream>>>(s_bf, wq_bf, bq, q_bf);
    gemm_logits_kernel<<<(B_ * (N_ / 16)) / 8, 256, 0, stream>>>(kbf, q_bf, logits);
    hipMemsetAsync(colsum, 0, (size_t)B_ * K_ * 4, stream);
    softmax_kernel<<<(B_ * N_) / 256, 256, 0, stream>>>(logits, out_attns, it, colsum);
    attn_norm_kernel<<<(B_ * N_) / 256, 256, 0, stream>>>(out_attns, it, colsum,
                                                          out_attn, attnT);
    gemm_updates_kernel<<<(B_ * 48) / 8, 256, 0, stream>>>(attnT, vbf, upd_bf);
    gemm_gru_kernel<<<(B_ * 144) / 8, 256, 0, stream>>>(upd_bf, slots_bf, wih_bf,
                                                        whh_bf, bih, bhh, gi, gh);
    gru_elem_kernel<<<nb(B_ * K_ * D_), 256, 0, stream>>>(gi, gh, slots_f, h_f);
    ln_rows_kernel<<<(B_ * K_) / 8, 256, 0, stream>>>(h_f, ln_mlp_g, ln_mlp_b, m_bf);
    gemm_mlp1_kernel<<<(B_ * 96) / 8, 256, 0, stream>>>(m_bf, w1_bf, b1, hid_bf);
    gemm_mlp2_kernel<<<(B_ * 48) / 8, 256, 0, stream>>>(hid_bf, w2_bf, b2, h_f,
                                                        slots_f, slots_bf);
  }
  hipMemcpyAsync(out_slots, slots_f, (size_t)B_ * K_ * D_ * 4,
                 hipMemcpyDeviceToDevice, stream);
}
